// TPConditionalFeedForward_59399397704334
// MI455X (gfx1250) — matrix operand order repaired and checked by reading, not compiled, not run on hardware
//
#include <hip/hip_runtime.h>
#include <math.h>

// Problem constants (match reference)
constexpr int Tt  = 2048;
constexpr int Ee  = 8;
constexpr int Dd  = 2048;
constexpr int Ii  = 5632;
constexpr int Aa  = 2;
constexpr int CAP = Tt * Aa;       // max entries per expert bucket

typedef __bf16 v16bf __attribute__((ext_vector_type(16)));
typedef float  v8f   __attribute__((ext_vector_type(8)));

union BFfrag {
    v16bf v;
    unsigned int u[8];
};

// fp32 pair -> packed bf16 dword (elem0 in low half), RNE in uint domain.
__device__ __forceinline__ unsigned int pack2bf(float a, float b) {
    unsigned int ua = __float_as_uint(a);
    unsigned int ub = __float_as_uint(b);
    ua += 0x7FFFu + ((ua >> 16) & 1u);
    ub += 0x7FFFu + ((ub >> 16) & 1u);
    return (ua >> 16) | (ub & 0xFFFF0000u);
}

__device__ __forceinline__ unsigned short f2bf1(float f) {
    unsigned int u = __float_as_uint(f);
    u += 0x7FFFu + ((u >> 16) & 1u);
    return (unsigned short)(u >> 16);
}

// silu(h1)*h3 with fast reciprocal (v_rcp_f32) — no IEEE div chain.
__device__ __forceinline__ float swiglu(float h1, float h3) {
    return h1 * __builtin_amdgcn_rcpf(1.0f + __expf(-h1)) * h3;
}

// ---------------- routing ----------------
__global__ void k_zero_counts(int* cnt) {
    if (threadIdx.x < Ee) cnt[threadIdx.x] = 0;
}

__global__ void k_route(const int* __restrict__ idx, int* __restrict__ cnt,
                        int* __restrict__ list) {
    int n = blockIdx.x * blockDim.x + threadIdx.x;   // pair id = t*A + a
    if (n < Tt * Aa) {
        int e = idx[n];
        int p = atomicAdd(&cnt[e], 1);
        list[e * CAP + p] = n;
    }
}

// ---------------- one-time fp32 -> bf16 conversion passes ----------------
__global__ void __launch_bounds__(256)
k_cvt(const float4* __restrict__ src, uint2* __restrict__ dst, int n4) {
    int i = blockIdx.x * 256 + threadIdx.x;
    if (i < n4) {
        float4 f = src[i];
        uint2 o;
        o.x = pack2bf(f.x, f.y);
        o.y = pack2bf(f.z, f.w);
        dst[i] = o;
    }
}

// Transpose-convert w2 [E][I][D] fp32 -> w2t [E][D][I] bf16 via 64x64 LDS tiles.
__global__ void __launch_bounds__(256)
k_cvt_w2t(const float* __restrict__ w2, unsigned short* __restrict__ w2t) {
    __shared__ unsigned short tls[64 * 72];    // [i][d] halves, pad 72
    const int e  = blockIdx.z;
    const int d0 = blockIdx.x * 64;
    const int k0 = blockIdx.y * 64;
    const float* src = w2 + ((size_t)e * Ii + k0) * Dd + d0;
    {   // read phase: thread r (i row 0..63) x 16 d, coalesced along d
        int r = threadIdx.x >> 2;
        int c = (threadIdx.x & 3) * 16;
        const float4* p = (const float4*)(src + (size_t)r * Dd + c);
        float4 f0 = p[0], f1 = p[1], f2 = p[2], f3 = p[3];
        unsigned int* dst = (unsigned int*)&tls[r * 72 + c];
        dst[0] = pack2bf(f0.x, f0.y); dst[1] = pack2bf(f0.z, f0.w);
        dst[2] = pack2bf(f1.x, f1.y); dst[3] = pack2bf(f1.z, f1.w);
        dst[4] = pack2bf(f2.x, f2.y); dst[5] = pack2bf(f2.z, f2.w);
        dst[6] = pack2bf(f3.x, f3.y); dst[7] = pack2bf(f3.z, f3.w);
    }
    __syncthreads();
    {   // write phase: thread dr (d row 0..63) x 16 i -> 32B contiguous store
        int dr = threadIdx.x >> 2;
        int is = (threadIdx.x & 3) * 16;
        const unsigned int* lds = (const unsigned int*)tls;
        unsigned int sel = (dr & 1) ? 0x07060302u : 0x05040100u;
        unsigned int* dst =
            (unsigned int*)(w2t + ((size_t)e * Dd + d0 + dr) * Ii + k0 + is);
#pragma unroll
        for (int j = 0; j < 8; j++) {
            unsigned int u0 = lds[(is + 2 * j)     * 36 + (dr >> 1)];
            unsigned int u1 = lds[(is + 2 * j + 1) * 36 + (dr >> 1)];
            dst[j] = __builtin_amdgcn_perm(u1, u0, sel);   // {u1.half, u0.half}
        }
    }
}

// ================= FAST PATH (pre-converted bf16 operands) =================

// GEMM1: h = silu(x@w1^T) * (x@w3^T). grid: (I/128, E), 256 thr (8 waves).
__global__ void __launch_bounds__(256)
k_gemm1_bf(const unsigned short* __restrict__ xb, const unsigned short* __restrict__ w1b,
           const unsigned short* __restrict__ w3b, const int* __restrict__ cnt,
           const int* __restrict__ list, unsigned short* __restrict__ hbuf) {
    __shared__ unsigned short xs[2][128 * 32];   // double-buffered X tile
    __shared__ int s_tok[128];

    const int e    = blockIdx.y;
    const int wave = threadIdx.x >> 5;
    const int lane = threadIdx.x & 31;
    const int lh   = lane >> 4;
    const int ln   = lane & 15;
    const int i0   = blockIdx.x * 128 + wave * 16;
    const int n_e  = cnt[e];
    const unsigned short* w1e = w1b + (size_t)e * Ii * Dd;
    const unsigned short* w3e = w3b + (size_t)e * Ii * Dd;

    const int sr = threadIdx.x >> 1;             // staging row
    const int sc = (threadIdx.x & 1) * 16;       // staging col

    for (int mbase = 0; mbase < n_e; mbase += 128) {
        if (threadIdx.x < 128) {
            int mi = mbase + threadIdx.x;
            s_tok[threadIdx.x] = list[e * CAP + (mi < n_e ? mi : n_e - 1)];
        }
        __syncthreads();

        auto stageX = [&](int buf, int kk) {
            const uint4* src =
                (const uint4*)(xb + (size_t)(s_tok[sr] >> 1) * Dd + kk + sc);
            uint4 v0 = src[0], v1 = src[1];
            uint4* dst = (uint4*)&xs[buf][sr * 32 + sc];
            dst[0] = v0; dst[1] = v1;
        };

        stageX(0, 0);                            // prime buffer 0

        v8f acc1[8] = {};
        v8f acc3[8] = {};
        int buf = 0;

        for (int k = 0; k < Dd; k += 32) {
            __syncthreads();                     // buffer `buf` visible

            // B fragments: 32B contiguous bf16 per lane, straight from global
            BFfrag b1, b3;
            {
                const unsigned int* p1 =
                    (const unsigned int*)(w1e + (size_t)(i0 + ln) * Dd + k + lh * 16);
                const unsigned int* p3 =
                    (const unsigned int*)(w3e + (size_t)(i0 + ln) * Dd + k + lh * 16);
                __builtin_prefetch(p1 + 16, 0, 1);   // next k-step
                __builtin_prefetch(p3 + 16, 0, 1);
#pragma unroll
                for (int j = 0; j < 8; j++) { b1.u[j] = p1[j]; b3.u[j] = p3[j]; }
            }

            if (k + 32 < Dd) stageX(buf ^ 1, k + 32);   // overlap next stage

#pragma unroll
            for (int m = 0; m < 8; m++) {
                BFfrag a;
                const unsigned int* src =
                    (const unsigned int*)&xs[buf][(m * 16 + ln) * 32];
#pragma unroll
                for (int p = 0; p < 8; p++) {
                    int K0 = 16 * lh + 2 * p;
                    a.u[p] = src[K0 >> 1];
                }
                acc1[m] = __builtin_amdgcn_wmma_f32_16x16x32_bf16(
                    false, a.v, false, b1.v, (short)0, acc1[m], false, false);
                acc3[m] = __builtin_amdgcn_wmma_f32_16x16x32_bf16(
                    false, a.v, false, b3.v, (short)0, acc3[m], false, false);
            }
            buf ^= 1;
        }

#pragma unroll
        for (int m = 0; m < 8; m++) {
#pragma unroll
            for (int r = 0; r < 8; r++) {
                int lrow = m * 16 + 8 * lh + r;
                if (mbase + lrow < n_e) {
                    float hv = swiglu(acc1[m][r], acc3[m][r]);
                    hbuf[(size_t)s_tok[lrow] * Ii + i0 + ln] = f2bf1(hv);
                }
            }
        }
        __syncthreads();
    }
}

// GEMM2: out = h @ w2 using transposed bf16 w2t[E][D][I].
__global__ void __launch_bounds__(256)
k_gemm2_bf(const unsigned short* __restrict__ hbuf, const unsigned short* __restrict__ w2t,
           const int* __restrict__ cnt, const int* __restrict__ list,
           float* __restrict__ out) {
    __shared__ unsigned short hs[2][128 * 32];   // double-buffered H tile
    __shared__ int s_tok[128];

    const int e    = blockIdx.y;
    const int wave = threadIdx.x >> 5;
    const int lane = threadIdx.x & 31;
    const int lh   = lane >> 4;
    const int ln   = lane & 15;
    const int dw   = blockIdx.x * 128 + wave * 16;
    const int n_e  = cnt[e];
    const unsigned short* w2e = w2t + (size_t)e * Dd * Ii;

    const int sr = threadIdx.x >> 1;
    const int sc = (threadIdx.x & 1) * 16;

    for (int mbase = 0; mbase < n_e; mbase += 128) {
        if (threadIdx.x < 128) {
            int mi = mbase + threadIdx.x;
            s_tok[threadIdx.x] = list[e * CAP + (mi < n_e ? mi : n_e - 1)];
        }
        __syncthreads();

        auto stageH = [&](int buf, int kk) {
            const uint4* src =
                (const uint4*)(hbuf + (size_t)s_tok[sr] * Ii + kk + sc);
            uint4 v0 = src[0], v1 = src[1];
            uint4* dst = (uint4*)&hs[buf][sr * 32 + sc];
            dst[0] = v0; dst[1] = v1;
        };

        stageH(0, 0);

        v8f acc[8] = {};
        int buf = 0;

        for (int k = 0; k < Ii; k += 32) {
            __syncthreads();

            BFfrag b;
            {
                const unsigned int* p =
                    (const unsigned int*)(w2e + (size_t)(dw + ln) * Ii + k + lh * 16);
                __builtin_prefetch(p + 16, 0, 1);
#pragma unroll
                for (int j = 0; j < 8; j++) b.u[j] = p[j];
            }

            if (k + 32 < Ii) stageH(buf ^ 1, k + 32);

#pragma unroll
            for (int m = 0; m < 8; m++) {
                BFfrag a;
                const unsigned int* src =
                    (const unsigned int*)&hs[buf][(m * 16 + ln) * 32];
#pragma unroll
                for (int p = 0; p < 8; p++) {
                    int K0 = 16 * lh + 2 * p;
                    a.u[p] = src[K0 >> 1];
                }
                acc[m] = __builtin_amdgcn_wmma_f32_16x16x32_bf16(
                    false, a.v, false, b.v, (short)0, acc[m], false, false);
            }
            buf ^= 1;
        }

#pragma unroll
        for (int m = 0; m < 8; m++) {
#pragma unroll
            for (int r = 0; r < 8; r++) {
                int lrow = m * 16 + 8 * lh + r;
                if (mbase + lrow < n_e) {
                    out[(size_t)s_tok[lrow] * Dd + dw + ln] = acc[m][r];
                }
            }
        }
        __syncthreads();
    }
}

// ================= FALLBACK PATH (convert in-flight, small ws) =============

__global__ void __launch_bounds__(256)
k_gemm1(const float* __restrict__ x, const float* __restrict__ w1,
        const float* __restrict__ w3, const int* __restrict__ cnt,
        const int* __restrict__ list, unsigned short* __restrict__ hbuf) {
    __shared__ unsigned short xs[128 * 32];
    __shared__ int s_tok[128];

    const int e    = blockIdx.y;
    const int wave = threadIdx.x >> 5;
    const int lane = threadIdx.x & 31;
    const int lh   = lane >> 4;
    const int ln   = lane & 15;
    const int i0   = blockIdx.x * 128 + wave * 16;
    const int n_e  = cnt[e];
    const float* w1e = w1 + (size_t)e * Ii * Dd;
    const float* w3e = w3 + (size_t)e * Ii * Dd;

    for (int mbase = 0; mbase < n_e; mbase += 128) {
        if (threadIdx.x < 128) {
            int mi = mbase + threadIdx.x;
            s_tok[threadIdx.x] = list[e * CAP + (mi < n_e ? mi : n_e - 1)];
        }
        __syncthreads();

        v8f acc1[8] = {};
        v8f acc3[8] = {};

        for (int k = 0; k < Dd; k += 32) {
            {
                int r = threadIdx.x >> 1;
                int c = (threadIdx.x & 1) * 16;
                const float* xr = x + (size_t)(s_tok[r] >> 1) * Dd + k + c;
                float4 f0 = ((const float4*)xr)[0];
                float4 f1 = ((const float4*)xr)[1];
                float4 f2 = ((const float4*)xr)[2];
                float4 f3 = ((const float4*)xr)[3];
                unsigned int* dst = (unsigned int*)&xs[r * 32 + c];
                dst[0] = pack2bf(f0.x, f0.y); dst[1] = pack2bf(f0.z, f0.w);
                dst[2] = pack2bf(f1.x, f1.y); dst[3] = pack2bf(f1.z, f1.w);
                dst[4] = pack2bf(f2.x, f2.y); dst[5] = pack2bf(f2.z, f2.w);
                dst[6] = pack2bf(f3.x, f3.y); dst[7] = pack2bf(f3.z, f3.w);
            }
            __syncthreads();

            BFfrag b1, b3;
            {
                const float* p1 = w1e + (size_t)(i0 + ln) * Dd + k + lh * 16;
                const float* p3 = w3e + (size_t)(i0 + ln) * Dd + k + lh * 16;
                __builtin_prefetch(p1 + 32, 0, 1);
                __builtin_prefetch(p3 + 32, 0, 1);
#pragma unroll
                for (int j = 0; j < 8; j++) {
                    float2 a = ((const float2*)p1)[j];
                    float2 c = ((const float2*)p3)[j];
                    b1.u[j] = pack2bf(a.x, a.y);
                    b3.u[j] = pack2bf(c.x, c.y);
                }
            }

#pragma unroll
            for (int m = 0; m < 8; m++) {
                BFfrag a;
                const unsigned int* src = (const unsigned int*)&xs[(m * 16 + ln) * 32];
#pragma unroll
                for (int p = 0; p < 8; p++) {
                    int K0 = 16 * lh + 2 * p;
                    a.u[p] = src[K0 >> 1];
                }
                acc1[m] = __builtin_amdgcn_wmma_f32_16x16x32_bf16(
                    false, a.v, false, b1.v, (short)0, acc1[m], false, false);
                acc3[m] = __builtin_amdgcn_wmma_f32_16x16x32_bf16(
                    false, a.v, false, b3.v, (short)0, acc3[m], false, false);
            }
            __syncthreads();
        }

#pragma unroll
        for (int m = 0; m < 8; m++) {
#pragma unroll
            for (int r = 0; r < 8; r++) {
                int lrow = m * 16 + 8 * lh + r;
                if (mbase + lrow < n_e) {
                    float hv = swiglu(acc1[m][r], acc3[m][r]);
                    hbuf[(size_t)s_tok[lrow] * Ii + i0 + ln] = f2bf1(hv);
                }
            }
        }
        __syncthreads();
    }
}

__global__ void __launch_bounds__(256)
k_gemm2(const unsigned short* __restrict__ hbuf, const float* __restrict__ w2,
        const int* __restrict__ cnt, const int* __restrict__ list,
        float* __restrict__ out) {
    __shared__ unsigned short hs[128 * 32];
    __shared__ unsigned short w2s[128 * 40];
    __shared__ int s_tok[128];

    const int e    = blockIdx.y;
    const int wave = threadIdx.x >> 5;
    const int lane = threadIdx.x & 31;
    const int lh   = lane >> 4;
    const int ln   = lane & 15;
    const int d0   = blockIdx.x * 128;
    const int dw   = d0 + wave * 16;
    const int n_e  = cnt[e];
    const float* w2e = w2 + (size_t)e * Ii * Dd;

    for (int mbase = 0; mbase < n_e; mbase += 128) {
        if (threadIdx.x < 128) {
            int mi = mbase + threadIdx.x;
            s_tok[threadIdx.x] = list[e * CAP + (mi < n_e ? mi : n_e - 1)];
        }
        __syncthreads();

        v8f acc[8] = {};

        for (int k = 0; k < Ii; k += 32) {
            {
                int r = threadIdx.x >> 1;
                int c = (threadIdx.x & 1) * 16;
                const uint4* src = (const uint4*)(hbuf + (size_t)s_tok[r] * Ii + k + c);
                uint4 v0 = src[0], v1 = src[1];
                uint4* dst = (uint4*)&hs[r * 32 + c];
                dst[0] = v0; dst[1] = v1;
            }
            {
                int c  = (threadIdx.x & 15) * 8;
                int rp = (threadIdx.x >> 4) * 2;
                const float* s0 = w2e + (size_t)(k + rp) * Dd + d0 + c;
                const float* s1 = s0 + Dd;
                __builtin_prefetch(s0 + 32 * Dd, 0, 1);
                float4 a0 = ((const float4*)s0)[0];
                float4 a1 = ((const float4*)s0)[1];
                float4 b0 = ((const float4*)s1)[0];
                float4 b1 = ((const float4*)s1)[1];
                float ra[8] = {a0.x, a0.y, a0.z, a0.w, a1.x, a1.y, a1.z, a1.w};
                float rb[8] = {b0.x, b0.y, b0.z, b0.w, b1.x, b1.y, b1.z, b1.w};
                unsigned int* dst = (unsigned int*)w2s;
#pragma unroll
                for (int j = 0; j < 8; j++)
                    dst[(c + j) * 20 + (rp >> 1)] = pack2bf(ra[j], rb[j]);
            }
            __syncthreads();

            BFfrag b;
            {
                const unsigned int* p =
                    (const unsigned int*)&w2s[(wave * 16 + ln) * 40 + lh * 16];
#pragma unroll
                for (int j = 0; j < 8; j++) b.u[j] = p[j];
            }

#pragma unroll
            for (int m = 0; m < 8; m++) {
                BFfrag a;
                const unsigned int* src = (const unsigned int*)&hs[(m * 16 + ln) * 32];
#pragma unroll
                for (int p = 0; p < 8; p++) {
                    int K0 = 16 * lh + 2 * p;
                    a.u[p] = src[K0 >> 1];
                }
                acc[m] = __builtin_amdgcn_wmma_f32_16x16x32_bf16(
                    false, a.v, false, b.v, (short)0, acc[m], false, false);
            }
            __syncthreads();
        }

#pragma unroll
        for (int m = 0; m < 8; m++) {
#pragma unroll
            for (int r = 0; r < 8; r++) {
                int lrow = m * 16 + 8 * lh + r;
                if (mbase + lrow < n_e) {
                    out[(size_t)s_tok[lrow] * Dd + dw + ln] = acc[m][r];
                }
            }
        }
        __syncthreads();
    }
}

// ---------------- launch ----------------
extern "C" void kernel_launch(void* const* d_in, const int* in_sizes, int n_in,
                              void* d_out, int out_size, void* d_ws, size_t ws_size,
                              hipStream_t stream) {
    const float* x  = (const float*)d_in[0];
    const int*   idx = (const int*)d_in[1];     // expert_indices [T, A]
    const float* w1 = (const float*)d_in[2];
    const float* w2 = (const float*)d_in[3];
    const float* w3 = (const float*)d_in[4];
    float* out = (float*)d_out;

    char* ws = (char*)d_ws;
    const size_t list_off  = 256;
    const size_t list_sz   = (size_t)Ee * CAP * 4;              // 128 KB
    const size_t hbuf_off  = list_off + list_sz;
    const size_t hbuf_sz   = (size_t)Tt * Aa * Ii * 2;          // ~46 MB
    const size_t wmat_sz   = (size_t)Ee * Ii * Dd * 2;          // ~184.5 MB each
    const size_t xb_sz     = (size_t)Tt * Dd * 2;               // 8 MB
    const size_t w1b_off   = hbuf_off + hbuf_sz;
    const size_t w3b_off   = w1b_off + wmat_sz;
    const size_t w2t_off   = w3b_off + wmat_sz;
    const size_t xb_off    = w2t_off + wmat_sz;
    const size_t need_full = xb_off + xb_sz;                    // ~608 MB

    int* cnt  = (int*)ws;
    int* list = (int*)(ws + list_off);
    unsigned short* hbuf = (unsigned short*)(ws + hbuf_off);

    k_zero_counts<<<1, 32, 0, stream>>>(cnt);
    k_route<<<(Tt * Aa + 255) / 256, 256, 0, stream>>>(idx, cnt, list);

    if (ws_size >= need_full) {
        unsigned short* w1b = (unsigned short*)(ws + w1b_off);
        unsigned short* w3b = (unsigned short*)(ws + w3b_off);
        unsigned short* w2t = (unsigned short*)(ws + w2t_off);
        unsigned short* xb  = (unsigned short*)(ws + xb_off);

        const int n4w = Ee * Ii * Dd / 4;   // float4 count per weight matrix
        const int n4x = Tt * Dd / 4;
        k_cvt<<<(n4w + 255) / 256, 256, 0, stream>>>((const float4*)w1, (uint2*)w1b, n4w);
        k_cvt<<<(n4w + 255) / 256, 256, 0, stream>>>((const float4*)w3, (uint2*)w3b, n4w);
        k_cvt<<<(n4x + 255) / 256, 256, 0, stream>>>((const float4*)x,  (uint2*)xb,  n4x);
        dim3 gt(Dd / 64, Ii / 64, Ee);      // 32 x 88 x 8
        k_cvt_w2t<<<gt, 256, 0, stream>>>(w2, w2t);

        dim3 g1(Ii / 128, Ee);
        k_gemm1_bf<<<g1, 256, 0, stream>>>(xb, w1b, w3b, cnt, list, hbuf);
        dim3 g2(Dd / 128, Ee);
        k_gemm2_bf<<<g2, 256, 0, stream>>>(hbuf, w2t, cnt, list, out);
    } else {
        dim3 g1(Ii / 128, Ee);
        k_gemm1<<<g1, 256, 0, stream>>>(x, w1, w3, cnt, list, hbuf);
        dim3 g2(Dd / 128, Ee);
        k_gemm2<<<g2, 256, 0, stream>>>(hbuf, w2, cnt, list, out);
    }

    (void)in_sizes; (void)n_in; (void)out_size;
}
